// CrossAttnBlock_64759516889865
// MI455X (gfx1250) — compile-verified, matrix operand order checked by reading, not run on hardware
//
#include <hip/hip_runtime.h>
#include <hip/hip_bf16.h>

// ---------------------------------------------------------------------------
// CDNA5 (gfx1250) cross-attention block, f16 WMMA everywhere, flash softmax.
// Fixed problem size from the reference: B=2, K=1024, S=4096, D=1024, H=16.
// Round 3: Tensor Data Mover (TENSOR_LOAD_TO_LDS + s_wait_tensorcnt) for K
// staging, async global->LDS (ASYNCcnt) for V staging, 64x64 register-blocked
// WMMA GEMMs.
// ---------------------------------------------------------------------------

typedef _Float16 half8  __attribute__((ext_vector_type(8)));
typedef _Float16 half16 __attribute__((ext_vector_type(16)));
typedef float    float8 __attribute__((ext_vector_type(8)));
typedef unsigned uint4v __attribute__((ext_vector_type(4)));
typedef unsigned uint8v __attribute__((ext_vector_type(8)));

__device__ __forceinline__ float8 zero8() {
  float8 z;
#pragma unroll
  for (int i = 0; i < 8; ++i) z[i] = 0.f;
  return z;
}

// A-fragment packing for V_WMMA_*_16X16X32_F16 (ISA 7.12.2, 16-bit A 16x32):
//   lanes 0-15  hold row M=lane,  elements [0..7]=K(k0..k0+7),  [8..15]=K(k0+16..k0+23)
//   lanes 16-31 hold row M=lane-16, K offset by +8 / +24.
// Caller passes p = row_ptr + k0 + (lane>>4)*8; we load 8 halfs at p and p+16.
__device__ __forceinline__ half16 load_frag(const _Float16* p) {
  half8 lo = *(const half8*)p;
  half8 hi = *(const half8*)(p + 16);
  half16 r;
#pragma unroll
  for (int i = 0; i < 8; ++i) { r[i] = lo[i]; r[8 + i] = hi[i]; }
  return r;
}

__device__ __forceinline__ float8 wmma16x16x32(half16 a, half16 b, float8 c) {
  // (neg_a, A, neg_b, B, c_mod, C, reuse_a, reuse_b)
  return __builtin_amdgcn_wmma_f32_16x16x32_f16(false, a, false, b, (short)0, c,
                                                false, false);
}

__device__ __forceinline__ float softplusf(float x) {
  return (x > 20.f) ? x : log1pf(expf(x));
}

// Async copy of 16 bytes global -> LDS, tracked by ASYNCcnt (ISA 15.18:
// GLOBAL_LOAD_ASYNC_TO_LDS_B128). LDS byte address is the low 32 bits of the
// flat shared-aperture pointer (ISA 10.2).
__device__ __forceinline__ void async_copy_b128(void* lds_ptr, const void* gptr) {
  unsigned lds_addr = (unsigned)(unsigned long long)lds_ptr;
  asm volatile("global_load_async_to_lds_b128 %0, %1, off"
               :: "v"(lds_addr), "v"(gptr) : "memory");
}
__device__ __forceinline__ void wait_asynccnt0() {
  asm volatile("s_wait_asynccnt 0" ::: "memory");
}

// Tensor Data Mover: DMA a 2D tile (tile_h rows x tile_w f16 elements, row
// stride `stride_elems`) from global memory into LDS with a single
// TENSOR_LOAD_TO_LDS.  Descriptor per ISA 8.3/8.4:
//   group0: count=1 | lds_addr | global_addr[56:0] | type=2
//   group1: data_size=1 (2B), tensor_dim0/1, tile_dim0/1, tensor_dim0_stride;
//           workgroup_mask=0 (not in a cluster), no pad/iterate/barrier.
// 2-group form (VADDR2/VADDR3 = NULL): tensors up to 2D.  Completion is
// tracked with TENSORcnt (wave-level; EXEC ignored).
__device__ __forceinline__ void tdm_load_2d_f16(void* lds_ptr, const void* gptr,
                                                unsigned tile_w, unsigned tile_h,
                                                unsigned stride_elems) {
  const unsigned long long ga = (unsigned long long)gptr;
  uint4v g0;
  g0[0] = 1u;                                           // count=1, user D#
  g0[1] = (unsigned)(unsigned long long)lds_ptr;        // lds_addr (bytes)
  g0[2] = (unsigned)(ga & 0xFFFFFFFFu);                 // global_addr[31:0]
  g0[3] = (unsigned)((ga >> 32) & 0x01FFFFFFu)          // global_addr[56:32]
        | (2u << 30);                                   // type=2 ("image")
  uint8v g1;
  g1[0] = (1u << 16);                                   // data_size=1 (2 bytes)
  g1[1] = (tile_w & 0xFFFFu) << 16;                     // tensor_dim0[15:0]
  g1[2] = (tile_w >> 16) | ((tile_h & 0xFFFFu) << 16);  // tdim0[31:16]|tdim1[15:0]
  g1[3] = (tile_h >> 16) | ((tile_w & 0xFFFFu) << 16);  // tdim1[31:16]|tile_dim0
  g1[4] = (tile_h & 0xFFFFu);                           // tile_dim1 (tile_dim2=0)
  g1[5] = stride_elems;                                 // tensor_dim0_stride[31:0]
  g1[6] = 0u;                                           // stride[47:32], tdim1_stride lo
  g1[7] = 0u;                                           // tdim1_stride hi (unused, 2D)
  asm volatile("tensor_load_to_lds %0, %1" :: "s"(g0), "s"(g1) : "memory");
}
__device__ __forceinline__ void wait_tensorcnt0() {
  __builtin_amdgcn_s_wait_tensorcnt(0);
}

// ---------------------------------------------------------------------------
// LayerNorm over rows of length Dm (f32 in, f16 out). One 256-thread block/row.
// ---------------------------------------------------------------------------
__global__ void __launch_bounds__(256)
layernorm_f16_kernel(const float* __restrict__ X, const float* __restrict__ w,
                     const float* __restrict__ b, _Float16* __restrict__ Y, int Dm) {
  const int row = blockIdx.x;
  const float* x = X + (size_t)row * Dm;
  float s = 0.f, s2 = 0.f;
  for (int i = threadIdx.x; i < Dm; i += blockDim.x) {
    float v = x[i];
    s += v; s2 += v * v;
  }
#pragma unroll
  for (int m = 16; m >= 1; m >>= 1) { s += __shfl_xor(s, m); s2 += __shfl_xor(s2, m); }
  __shared__ float rs[8], rs2[8];
  const int wid = threadIdx.x >> 5, lane = threadIdx.x & 31;
  if (lane == 0) { rs[wid] = s; rs2[wid] = s2; }
  __syncthreads();
  if (wid == 0) {
    s  = (lane < 8) ? rs[lane]  : 0.f;
    s2 = (lane < 8) ? rs2[lane] : 0.f;
#pragma unroll
    for (int m = 4; m >= 1; m >>= 1) { s += __shfl_xor(s, m); s2 += __shfl_xor(s2, m); }
    if (lane == 0) { rs[0] = s; rs2[0] = s2; }
  }
  __syncthreads();
  const float mu  = rs[0] / (float)Dm;
  const float var = rs2[0] / (float)Dm - mu * mu;
  const float inv = rsqrtf(var + 1e-5f);
  for (int i = threadIdx.x; i < Dm; i += blockDim.x)
    Y[(size_t)row * Dm + i] = (_Float16)((x[i] - mu) * inv * w[i] + b[i]);
}

// ---------------------------------------------------------------------------
// W [Kd, N] f32 (row-major) -> Wt [N, Kd] f16 (row-major, i.e. W transposed)
// ---------------------------------------------------------------------------
__global__ void __launch_bounds__(256)
transpose_f32_to_f16_kernel(const float* __restrict__ W, _Float16* __restrict__ Wt,
                            int Kd, int N) {
  const int idx = blockIdx.x * blockDim.x + threadIdx.x;
  if (idx >= Kd * N) return;
  const int n = idx / Kd, k = idx % Kd;          // coalesced writes along k
  Wt[(size_t)n * Kd + k] = (_Float16)W[(size_t)k * N + n];
}

// ---------------------------------------------------------------------------
// GEMM: C[M,N] = A[M,Kd](f16) x Wt[N,Kd](f16, pre-transposed) + bias.
// One wave computes a 64x64 macro-tile = 4x4 WMMA tiles; every A/B fragment
// is reused 4x (16 WMMAs per 8 fragment loads per k-step).
// Requires M % 64 == 0 and N % 64 == 0 (true for all call sites here).
// MODE 0: f16 store.  MODE 1: f32 store + residual add.  MODE 2: f16 + GELU.
// ---------------------------------------------------------------------------
template <int MODE>
__global__ void __launch_bounds__(256)
gemm_wmma_kernel(const _Float16* __restrict__ A, const _Float16* __restrict__ Bt,
                 const float* __restrict__ bias, const float* __restrict__ resid,
                 void* __restrict__ Cout, int M, int N, int Kd) {
  const int wave    = blockIdx.x * (blockDim.x >> 5) + (threadIdx.x >> 5);
  const int tiles_n = N >> 6;
  const int total   = (M >> 6) * tiles_n;
  if (wave >= total) return;                     // wave-uniform; EXEC stays all-1s
  const int lane = threadIdx.x & 31;
  const int hf   = lane >> 4;
  const int l    = lane & 15;
  const int tm   = wave / tiles_n, tn = wave % tiles_n;

  const _Float16* abase = A  + (size_t)(tm * 64 + l) * Kd + hf * 8;
  const _Float16* bbase = Bt + (size_t)(tn * 64 + l) * Kd + hf * 8;
  const size_t    rstep = (size_t)16 * Kd;       // 16 rows down

  float8 acc[4][4];
#pragma unroll
  for (int rt = 0; rt < 4; ++rt)
#pragma unroll
    for (int ct = 0; ct < 4; ++ct) acc[rt][ct] = zero8();

  for (int k0 = 0; k0 < Kd; k0 += 32) {
    __builtin_prefetch((const void*)(abase + k0 + 256), 0, 1);  // global_prefetch_b8
    __builtin_prefetch((const void*)(bbase + k0 + 256), 0, 1);
    half16 af[4], bf[4];
#pragma unroll
    for (int t = 0; t < 4; ++t) {
      af[t] = load_frag(abase + t * rstep + k0);
      bf[t] = load_frag(bbase + t * rstep + k0);
    }
#pragma unroll
    for (int rt = 0; rt < 4; ++rt)
#pragma unroll
      for (int ct = 0; ct < 4; ++ct)
        acc[rt][ct] = wmma16x16x32(af[rt], bf[ct], acc[rt][ct]);
  }

  // C layout (ISA 7.12.2): lanes 0-15 -> N=lane, M=vgpr; lanes 16-31 -> M=vgpr+8
#pragma unroll
  for (int ct = 0; ct < 4; ++ct) {
    const int   col = tn * 64 + ct * 16 + l;
    const float bv  = bias[col];
#pragma unroll
    for (int rt = 0; rt < 4; ++rt) {
#pragma unroll
      for (int i = 0; i < 8; ++i) {
        const int row = tm * 64 + rt * 16 + hf * 8 + i;
        float v = acc[rt][ct][i] + bv;
        if (MODE == 1) {
          ((float*)Cout)[(size_t)row * N + col] = v + resid[(size_t)row * N + col];
        } else {
          if (MODE == 2) v = 0.5f * v * (1.0f + erff(v * 0.70710678118654752f));
          ((_Float16*)Cout)[(size_t)row * N + col] = (_Float16)v;
        }
      }
    }
  }
}

// ---------------------------------------------------------------------------
// Flash attention: one wave per (b, h, 16-row Q tile).  K chunk (32x64 f16)
// staged in LDS via the Tensor Data Mover (one TENSOR_LOAD_TO_LDS per chunk,
// TENSORcnt); V chunk staged via GLOBAL_LOAD_ASYNC_TO_LDS_B128 (ASYNCcnt).
// Scores via WMMA; online softmax with bias/gate fusion; P x V via WMMA.
// Grid sized exactly: no wave exits early (syncthreads in S loop).
// ---------------------------------------------------------------------------
__global__ void __launch_bounds__(128)
flash_attn_kernel(const _Float16* __restrict__ qp, const _Float16* __restrict__ kp,
                  const _Float16* __restrict__ vp,
                  const float* __restrict__ attn_bias, const float* __restrict__ obs_bias,
                  const float* __restrict__ density,
                  const float* __restrict__ dist_raw, const float* __restrict__ obs_raw,
                  const float* __restrict__ dens_raw,
                  _Float16* __restrict__ out,
                  int Kk, int Ss, int Hh, int Dd) {
  __shared__ __align__(16) _Float16 ldsK[4][32 * 64];
  __shared__ __align__(16) _Float16 ldsV[4][32 * 64];
  __shared__ __align__(16) _Float16 ldsP[4][16 * 32];

  const int wid    = threadIdx.x >> 5;
  const int lane   = threadIdx.x & 31;
  const int wave   = blockIdx.x * (blockDim.x >> 5) + wid;
  const int ktiles = Kk >> 4;
  const int kt = wave % ktiles;
  const int bh = wave / ktiles;
  const int h  = bh % Hh;
  const int b  = bh / Hh;
  const int hf = lane >> 4;
  const int l  = lane & 15;

  _Float16* lk = ldsK[wid];
  _Float16* lv = ldsV[wid];
  _Float16* lp = ldsP[wid];

  const float dw    = softplusf(dist_raw[0]);
  const float ow    = softplusf(obs_raw[0]);
  const float tg    = tanhf(dens_raw[0]);
  const float scale = 0.125f;  // HD^-0.5, HD=64

  const _Float16* qrow = qp + ((size_t)b * Kk + kt * 16 + l) * Dd + h * 64 + hf * 8;
  const half16 qf0 = load_frag(qrow);
  const half16 qf1 = load_frag(qrow + 32);

  float gate[8], rowm[8], rowl[8];
#pragma unroll
  for (int i = 0; i < 8; ++i) {
    gate[i] = 1.f + tg * density[b * Kk + kt * 16 + hf * 8 + i];
    rowm[i] = -3.0e38f;
    rowl[i] = 0.f;
  }
  float8 o0 = zero8(), o1 = zero8(), o2 = zero8(), o3 = zero8();

  const float*    ab    = attn_bias + ((size_t)b * Kk + kt * 16) * Ss;
  const float*    ob    = obs_bias  + ((size_t)b * Kk + kt * 16) * Ss;
  const _Float16* kbase = kp + (size_t)b * Ss * Dd + h * 64;
  const _Float16* vbase = vp + (size_t)b * Ss * Dd + h * 64;

  for (int s0 = 0; s0 < Ss; s0 += 32) {
    // K chunk (32 rows x 64 halfs, row stride Dd): one TDM descriptor, one
    // instruction, DMA engine does the address generation.
    tdm_load_2d_f16(lk, kbase + (size_t)s0 * Dd, /*tile_w=*/64, /*tile_h=*/32,
                    /*stride_elems=*/(unsigned)Dd);
    // V chunk via async loads: 16B packets, 8 per lane.
#pragma unroll
    for (int t = 0; t < 8; ++t) {
      const int idx = t * 32 + lane;             // 0..255
      const int r = idx >> 3, pp = idx & 7;
      async_copy_b128(lv + r * 64 + pp * 8, vbase + (size_t)(s0 + r) * Dd + pp * 8);
    }
    wait_tensorcnt0();
    wait_asynccnt0();
    __syncthreads();

    // Scores: two 16x16 tiles (cols s0..s0+15, s0+16..s0+31), K=64 in 2 steps.
    float8 c0 = zero8(), c1 = zero8();
    {
      const _Float16* kr0 = lk + (size_t)l * 64 + hf * 8;
      c0 = wmma16x16x32(qf0, load_frag(kr0), c0);
      c0 = wmma16x16x32(qf1, load_frag(kr0 + 32), c0);
      const _Float16* kr1 = lk + (size_t)(16 + l) * 64 + hf * 8;
      c1 = wmma16x16x32(qf0, load_frag(kr1), c1);
      c1 = wmma16x16x32(qf1, load_frag(kr1 + 32), c1);
    }

    // Bias + gate + online softmax.  Row m lives at vgpr i of one 16-lane half,
    // so row reductions are xor-shuffles with masks 1,2,4,8 (wave32).
#pragma unroll
    for (int i = 0; i < 8; ++i) {
      const int r = hf * 8 + i;
      const float a0 = ab[(size_t)r * Ss + s0 + l];
      const float a1 = ab[(size_t)r * Ss + s0 + 16 + l];
      const float b0 = ob[(size_t)r * Ss + s0 + l];
      const float b1 = ob[(size_t)r * Ss + s0 + 16 + l];
      const float v0 = (c0[i] * scale + dw * a0 + ow * b0) * gate[i];
      const float v1 = (c1[i] * scale + dw * a1 + ow * b1) * gate[i];
      float mx = fmaxf(v0, v1);
#pragma unroll
      for (int m = 1; m <= 8; m <<= 1) mx = fmaxf(mx, __shfl_xor(mx, m));
      const float nm    = fmaxf(rowm[i], mx);
      const float alpha = __expf(rowm[i] - nm);
      const float p0    = __expf(v0 - nm);
      const float p1    = __expf(v1 - nm);
      float rsum = p0 + p1;
#pragma unroll
      for (int m = 1; m <= 8; m <<= 1) rsum += __shfl_xor(rsum, m);
      rowl[i] = rowl[i] * alpha + rsum;
      rowm[i] = nm;
      o0[i] *= alpha; o1[i] *= alpha; o2[i] *= alpha; o3[i] *= alpha;
      lp[r * 32 + l]      = (_Float16)p0;
      lp[r * 32 + 16 + l] = (_Float16)p1;
    }
    __syncthreads();

    // O += P(16x32) x V(32x64): A-frag from LDS P, B-frag = V columns from LDS.
    half16 pf;
    {
      const _Float16* pr = lp + l * 32 + hf * 8;
#pragma unroll
      for (int i = 0; i < 8; ++i) { pf[i] = pr[i]; pf[8 + i] = pr[16 + i]; }
    }
#pragma unroll
    for (int t = 0; t < 4; ++t) {
      half16 vf;
      const _Float16* vc = lv + (size_t)(hf * 8) * 64 + t * 16 + l;
#pragma unroll
      for (int i = 0; i < 8; ++i) {
        vf[i]     = vc[(size_t)i * 64];
        vf[8 + i] = vc[(size_t)(16 + i) * 64];
      }
      if      (t == 0) o0 = wmma16x16x32(pf, vf, o0);
      else if (t == 1) o1 = wmma16x16x32(pf, vf, o1);
      else if (t == 2) o2 = wmma16x16x32(pf, vf, o2);
      else             o3 = wmma16x16x32(pf, vf, o3);
    }
    __syncthreads();
  }

  _Float16* outb = out + ((size_t)b * Kk + kt * 16) * Dd + h * 64;
#pragma unroll
  for (int i = 0; i < 8; ++i) {
    const int   r   = hf * 8 + i;
    const float inv = 1.f / rowl[i];
    outb[(size_t)r * Dd +  0 + l] = (_Float16)(o0[i] * inv);
    outb[(size_t)r * Dd + 16 + l] = (_Float16)(o1[i] * inv);
    outb[(size_t)r * Dd + 32 + l] = (_Float16)(o2[i] * inv);
    outb[(size_t)r * Dd + 48 + l] = (_Float16)(o3[i] * inv);
  }
}

// ---------------------------------------------------------------------------
// Host-side orchestration.
// ---------------------------------------------------------------------------
extern "C" void kernel_launch(void* const* d_in, const int* in_sizes, int n_in,
                              void* d_out, int out_size, void* d_ws, size_t ws_size,
                              hipStream_t stream) {
  (void)in_sizes; (void)n_in; (void)out_size; (void)ws_size;
  constexpr int B = 2, K = 1024, S = 4096, D = 1024, H = 16;

  const float* q         = (const float*)d_in[0];
  const float* kv        = (const float*)d_in[1];
  const float* attn_bias = (const float*)d_in[2];
  const float* obs_bias  = (const float*)d_in[3];
  const float* density   = (const float*)d_in[4];
  const float* ln_q_w    = (const float*)d_in[5];
  const float* ln_q_b    = (const float*)d_in[6];
  const float* ln_kv_w   = (const float*)d_in[7];
  const float* ln_kv_b   = (const float*)d_in[8];
  const float* wq        = (const float*)d_in[9];
  const float* bq        = (const float*)d_in[10];
  const float* wk        = (const float*)d_in[11];
  const float* bk        = (const float*)d_in[12];
  const float* wv        = (const float*)d_in[13];
  const float* bv        = (const float*)d_in[14];
  const float* wo        = (const float*)d_in[15];
  const float* bo        = (const float*)d_in[16];
  const float* dist_raw  = (const float*)d_in[17];
  const float* obs_raw   = (const float*)d_in[18];
  const float* dens_raw  = (const float*)d_in[19];
  const float* ln_f_w    = (const float*)d_in[20];
  const float* ln_f_b    = (const float*)d_in[21];
  const float* w1        = (const float*)d_in[22];
  const float* b1        = (const float*)d_in[23];
  const float* w2        = (const float*)d_in[24];
  const float* b2        = (const float*)d_in[25];

  size_t off = 0;
  auto take = [&](size_t bytes) -> void* {
    void* p = (char*)d_ws + off;
    off += (bytes + 255) & ~(size_t)255;
    return p;
  };
  _Float16* q_ln  = (_Float16*)take((size_t)B * K * D * 2);
  _Float16* kv_ln = (_Float16*)take((size_t)B * S * D * 2);
  _Float16* wq_t  = (_Float16*)take((size_t)D * D * 2);
  _Float16* wk_t  = (_Float16*)take((size_t)D * D * 2);
  _Float16* wv_t  = (_Float16*)take((size_t)D * D * 2);
  _Float16* wo_t  = (_Float16*)take((size_t)D * D * 2);
  _Float16* w1_t  = (_Float16*)take((size_t)D * 2 * D * 2);
  _Float16* w2_t  = (_Float16*)take((size_t)2 * D * D * 2);
  _Float16* qp_h  = (_Float16*)take((size_t)B * K * D * 2);
  _Float16* kp_h  = (_Float16*)take((size_t)B * S * D * 2);
  _Float16* vp_h  = (_Float16*)take((size_t)B * S * D * 2);
  _Float16* ao_h  = (_Float16*)take((size_t)B * K * D * 2);
  float*    x_f32 = (float*)   take((size_t)B * K * D * 4);
  _Float16* x_ln  = (_Float16*)take((size_t)B * K * D * 2);
  _Float16* h_h   = (_Float16*)take((size_t)B * K * 2 * D * 2);

  // 64x64 macro-tiles, 8 waves (256 threads) per block
  auto gemm_blocks = [](int M, int N) { return ((M / 64) * (N / 64) + 7) / 8; };

  // 1) LayerNorms -> f16
  layernorm_f16_kernel<<<B * K, 256, 0, stream>>>(q,  ln_q_w,  ln_q_b,  q_ln,  D);
  layernorm_f16_kernel<<<B * S, 256, 0, stream>>>(kv, ln_kv_w, ln_kv_b, kv_ln, D);

  // 2) Weight transposes -> f16
  transpose_f32_to_f16_kernel<<<(D * D + 255) / 256, 256, 0, stream>>>(wq, wq_t, D, D);
  transpose_f32_to_f16_kernel<<<(D * D + 255) / 256, 256, 0, stream>>>(wk, wk_t, D, D);
  transpose_f32_to_f16_kernel<<<(D * D + 255) / 256, 256, 0, stream>>>(wv, wv_t, D, D);
  transpose_f32_to_f16_kernel<<<(D * D + 255) / 256, 256, 0, stream>>>(wo, wo_t, D, D);
  transpose_f32_to_f16_kernel<<<(D * 2 * D + 255) / 256, 256, 0, stream>>>(w1, w1_t, D, 2 * D);
  transpose_f32_to_f16_kernel<<<(2 * D * D + 255) / 256, 256, 0, stream>>>(w2, w2_t, 2 * D, D);

  // 3) Q/K/V projections (WMMA GEMMs, f16 out; [B*rows, D] == [B,rows,H,HD])
  gemm_wmma_kernel<0><<<gemm_blocks(B * K, D), 256, 0, stream>>>(
      q_ln, wq_t, bq, nullptr, qp_h, B * K, D, D);
  gemm_wmma_kernel<0><<<gemm_blocks(B * S, D), 256, 0, stream>>>(
      kv_ln, wk_t, bk, nullptr, kp_h, B * S, D, D);
  gemm_wmma_kernel<0><<<gemm_blocks(B * S, D), 256, 0, stream>>>(
      kv_ln, wv_t, bv, nullptr, vp_h, B * S, D, D);

  // 4) Flash attention: B*H*(K/16) = 2048 waves, 4 waves/block -> exactly 512 blocks
  flash_attn_kernel<<<(B * H * (K / 16)) / 4, 128, 0, stream>>>(
      qp_h, kp_h, vp_h, attn_bias, obs_bias, density,
      dist_raw, obs_raw, dens_raw, ao_h, K, S, H, D);

  // 5) Output projection + residual (x = q + attn @ wo + bo), f32
  gemm_wmma_kernel<1><<<gemm_blocks(B * K, D), 256, 0, stream>>>(
      ao_h, wo_t, bo, q, x_f32, B * K, D, D);

  // 6) Final LN + MLP
  layernorm_f16_kernel<<<B * K, 256, 0, stream>>>(x_f32, ln_f_w, ln_f_b, x_ln, D);
  gemm_wmma_kernel<2><<<gemm_blocks(B * K, 2 * D), 256, 0, stream>>>(
      x_ln, w1_t, b1, nullptr, h_h, B * K, 2 * D, D);
  gemm_wmma_kernel<1><<<gemm_blocks(B * K, D), 256, 0, stream>>>(
      h_h, w2_t, b2, x_f32, (float*)d_out, B * K, D, 2 * D);
}
